// CrossMambaInteraction_73315091743055
// MI455X (gfx1250) — compile-verified, matrix operand order checked by reading, not run on hardware
//
#include <hip/hip_runtime.h>
#include <hip/hip_bf16.h>
#include <math.h>

typedef __attribute__((ext_vector_type(16))) __bf16 v16bf;
typedef __attribute__((ext_vector_type(4)))  __bf16 v4bf;
typedef __attribute__((ext_vector_type(4)))  float  v4f;
typedef __attribute__((ext_vector_type(8)))  float  v8f;

#define BM 128
#define BN 64
#define BK 32
#define KPAD 4   // row pitch = 36 elements = 72B (8B aligned for packed b64 stores)

struct GemmP {
    const float* A;  long long a_batch; int lda; int a_mode; int a_roff; // 0 row-major, 1 col-major(A[k*lda+m]), 2 row-major w/ pad-remap rows
    const float* Bw; int ldb; int kstr;                                  // weight element at n*ldb + k*kstr + tap
    const float* bias;
    float* Out; long long o_batch; int ldo; int out_mode;                // 0 [m][n], 1 transposed [n][m], 2 pad-remap rows
    int K; int beta; int ntaps;
};

// float4 -> packed bf16x4 (lowers to paired v_cvt_pk_bf16_f32) + single ds_store_b64
__device__ __forceinline__ void store4_bf16(__bf16* dst, float4 v) {
    v4f f = { v.x, v.y, v.z, v.w };
    *(v4bf*)dst = __builtin_convertvector(f, v4bf);
}

struct TileRegs { float4 a[4]; float4 b[2]; };

__global__ __launch_bounds__(256)
void gemm_wmma_bf16(GemmP p) {
    __shared__ __bf16 As[BM][BK + KPAD];
    __shared__ __bf16 Bs[BN][BK + KPAD];

    const int tid  = threadIdx.x;
    const int bat  = blockIdx.z;
    const float* A = p.A + (long long)bat * p.a_batch;
    float* Out     = p.Out + (long long)bat * p.o_batch;
    const int t0 = blockIdx.x * BM;
    const int n0 = blockIdx.y * BN;
    const int wave = tid >> 5, lane = tid & 31;
    const int wm = wave >> 1, wn = wave & 1;
    const int hl = lane >> 4, li = lane & 15;

    // staging indices (fixed per thread)
    const int m4  = (tid & 31) * 4;   // a_mode 1
    const int kb  = tid >> 5;         // a_mode 1
    const int k4  = (tid & 7) * 4;    // a_mode 0/2 + B kstr==1
    const int mr  = tid >> 3;         // a_mode 0/2
    const int nr  = tid >> 3;         // B kstr==1
    const int kx  = tid & 31;         // B kstr!=1
    const int ny  = tid >> 5;         // B kstr!=1

    const int kpt    = p.K / BK;
    const int nsteps = p.ntaps * kpt; // always even for our shapes

    // step -> (k0, A row offset, B tap offset)
    auto stepOff = [&](int step, int& k0, int& aro, int& boff) {
        if (p.ntaps > 1) {            // 3x3 conv: kpt == 8
            int tap = step >> 3;
            k0   = (step & 7) * BK;
            aro  = (tap / 3) * 34 + (tap % 3);
            boff = tap;
        } else {
            k0 = step * BK; aro = p.a_roff; boff = 0;
        }
    };

    auto loadTile = [&](int step, TileRegs& r) {
        int k0, aro, boff; stepOff(step, k0, aro, boff);
        if (p.a_mode == 1) {
            #pragma unroll
            for (int i = 0; i < 4; ++i) {
                int k = kb + i * 8;
                r.a[i] = *(const float4*)&A[(long long)(k0 + k) * p.lda + (t0 + m4)];
            }
        } else {
            #pragma unroll
            for (int i = 0; i < 4; ++i) {
                int m = mr + i * 32;
                int t = t0 + m;
                int row = (p.a_mode == 2) ? (t + 2 * (t >> 5) + aro) : t;
                r.a[i] = *(const float4*)&A[(long long)row * p.lda + (k0 + k4)];
            }
        }
        if (p.kstr == 1) {
            #pragma unroll
            for (int i = 0; i < 2; ++i) {
                int n = nr + i * 32;
                r.b[i] = *(const float4*)&p.Bw[(long long)(n0 + n) * p.ldb + (k0 + k4)];
            }
        } else {
            float* rb = (float*)r.b;
            #pragma unroll
            for (int i = 0; i < 8; ++i) {
                int n = ny + i * 8;
                rb[i] = p.Bw[(long long)(n0 + n) * p.ldb + (long long)(k0 + kx) * p.kstr + boff];
            }
        }
    };

    auto storeTile = [&](const TileRegs& r) {
        if (p.a_mode == 1) {
            #pragma unroll
            for (int i = 0; i < 4; ++i) {
                int k = kb + i * 8;
                As[m4 + 0][k] = (__bf16)r.a[i].x;
                As[m4 + 1][k] = (__bf16)r.a[i].y;
                As[m4 + 2][k] = (__bf16)r.a[i].z;
                As[m4 + 3][k] = (__bf16)r.a[i].w;
            }
        } else {
            #pragma unroll
            for (int i = 0; i < 4; ++i)
                store4_bf16(&As[mr + i * 32][k4], r.a[i]);
        }
        if (p.kstr == 1) {
            #pragma unroll
            for (int i = 0; i < 2; ++i)
                store4_bf16(&Bs[nr + i * 32][k4], r.b[i]);
        } else {
            const float* rb = (const float*)r.b;
            #pragma unroll
            for (int i = 0; i < 8; ++i) Bs[ny + i * 8][kx] = (__bf16)rb[i];
        }
    };

    v8f acc[2][2] = {};

    auto computeTile = [&]() {
        v16bf fa[2], fb[2];
        #pragma unroll
        for (int f = 0; f < 2; ++f) {
            int m = wm * 32 + f * 16 + li;
            int n = wn * 32 + f * 16 + li;
            #pragma unroll
            for (int v = 0; v < 8; ++v) {
                int kk = ((v & 3) * 2) + ((v >> 2) * 16) + hl * 8;
                fa[f][2 * v]     = As[m][kk];
                fa[f][2 * v + 1] = As[m][kk + 1];
                fb[f][2 * v]     = Bs[n][kk];
                fb[f][2 * v + 1] = Bs[n][kk + 1];
            }
        }
        #pragma unroll
        for (int fm = 0; fm < 2; ++fm)
            #pragma unroll
            for (int fn = 0; fn < 2; ++fn)
                acc[fm][fn] = __builtin_amdgcn_wmma_f32_16x16x32_bf16(
                    false, fa[fm], false, fb[fn], (short)0, acc[fm][fn], false, false);
    };

    // software pipeline: register ping-pong, loads overlap compute
    TileRegs r0, r1;
    loadTile(0, r0);
    for (int step = 0; step < nsteps; step += 2) {
        __syncthreads();
        storeTile(r0);
        __syncthreads();
        if (step + 1 < nsteps) loadTile(step + 1, r1);
        computeTile();
        __syncthreads();
        storeTile(r1);
        __syncthreads();
        if (step + 2 < nsteps) loadTile(step + 2, r0);
        computeTile();
    }

    // ---- epilogue: C/D layout -> lane(li)=col, row = r + hl*8 ----
    #pragma unroll
    for (int fm = 0; fm < 2; ++fm)
        #pragma unroll
        for (int fn = 0; fn < 2; ++fn)
            #pragma unroll
            for (int r = 0; r < 8; ++r) {
                int gm = t0 + wm * 32 + fm * 16 + hl * 8 + r;
                int gn = n0 + wn * 32 + fn * 16 + li;
                float val = acc[fm][fn][r];
                if (p.bias) val += p.bias[gn];
                long long idx;
                if (p.out_mode == 1)      idx = (long long)gn * p.ldo + gm;
                else if (p.out_mode == 2) idx = (long long)(gm + 2 * (gm >> 5) + 35) * p.ldo + gn;
                else                      idx = (long long)gm * p.ldo + gn;
                if (p.beta) Out[idx] += val; else Out[idx] = val;
            }
}

__global__ void zero_kernel(float* p, long long n) {
    long long i = (long long)blockIdx.x * blockDim.x + threadIdx.x;
    long long s = (long long)gridDim.x * blockDim.x;
    for (; i < n; i += s) p[i] = 0.f;
}

// LayerNorm(channel) + SiLU in place on padded (B,1156,256) buffer; one wave per pixel.
__global__ __launch_bounds__(256)
void ln_silu_kernel(float* zlnp, const float* g, const float* bt) {
    int wave = threadIdx.x >> 5, lane = threadIdx.x & 31;
    int p = blockIdx.x * 8 + wave;          // pixel id 0..16383
    int b = p >> 10, t = p & 1023;
    int row = t + 2 * (t >> 5) + 35;
    float* ptr = zlnp + ((long long)b * 1156 + row) * 256;
    float x[8], s1 = 0.f, s2 = 0.f;
    #pragma unroll
    for (int i = 0; i < 8; ++i) { x[i] = ptr[lane + 32 * i]; s1 += x[i]; s2 += x[i] * x[i]; }
    #pragma unroll
    for (int o = 16; o > 0; o >>= 1) { s1 += __shfl_xor(s1, o); s2 += __shfl_xor(s2, o); }
    float mean = s1 * (1.f / 256.f);
    float var  = s2 * (1.f / 256.f) - mean * mean;
    float rstd = rsqrtf(var + 1e-5f);
    #pragma unroll
    for (int i = 0; i < 8; ++i) {
        int c = lane + 32 * i;
        float v = (x[i] - mean) * rstd * g[c] + bt[c];
        ptr[c] = v / (1.f + expf(-v));      // SiLU
    }
}

__global__ __launch_bounds__(256)
void pool_kernel(const float* z3, float* pooled) {   // grid(B), block 256 (=D)
    int b = blockIdx.x, d = threadIdx.x;
    const float* p = z3 + (long long)b * 1024 * 256 + d;
    float acc = 0.f;
    for (int t = 0; t < 1024; ++t) acc += p[(long long)t * 256];
    pooled[b * 256 + d] = acc * (1.f / 1024.f);
}

__global__ __launch_bounds__(128)
void head_kernel(const float* pooled, const float* w1, const float* b1,
                 const float* w2, const float* b2, const float* bar,
                 float* out, int isA) {              // grid(B), block 128
    __shared__ float t1[128];
    int b = blockIdx.x, j = threadIdx.x;
    const float* pv = pooled + b * 256;
    float s = b1[j];
    for (int k = 0; k < 256; ++k) s += pv[k] * w1[j * 256 + k];
    t1[j] = s / (1.f + expf(-s));
    __syncthreads();
    float o = b2[j];
    for (int k = 0; k < 128; ++k) o += t1[k] * w2[j * 128 + k];
    o *= bar[j];
    if (isA) { o = fminf(fmaxf(o, -20.f), 0.f); o = expf(o); }
    out[b * 128 + j] = o;
}

__global__ void scan_kernel(const float* u, const float* Av, const float* Bv, float* h) {
    int id = blockIdx.x * blockDim.x + threadIdx.x;  // B*N = 2048
    int b = id >> 7, n = id & 127;
    float a = Av[id], bc = Bv[id], hv = 0.f;
    const float* up = u + (long long)b * 1024 * 128 + n;
    float* hp       = h + (long long)b * 1024 * 128 + n;
    for (int t = 0; t < 1024; ++t) { hv = a * hv + bc * up[t * 128]; hp[t * 128] = hv; }
}

// channel-LN over C=512 in place on NCHW output; one thread per pixel.
__global__ __launch_bounds__(256)
void ln_out_kernel(float* out, const float* g, const float* bt) {
    int t = blockIdx.x * 256 + threadIdx.x;
    int b = blockIdx.y;
    float* ptr = out + (long long)b * 512 * 1024 + t;
    float s1 = 0.f, s2 = 0.f;
    for (int c = 0; c < 512; ++c) { float x = ptr[(long long)c * 1024]; s1 += x; s2 += x * x; }
    float mean = s1 * (1.f / 512.f);
    float var  = s2 * (1.f / 512.f) - mean * mean;
    float rstd = rsqrtf(var + 1e-5f);
    for (int c = 0; c < 512; ++c) {
        long long o = (long long)c * 1024;
        ptr[o] = (ptr[o] - mean) * rstd * g[c] + bt[c];
    }
}

extern "C" void kernel_launch(void* const* d_in, const int* in_sizes, int n_in,
                              void* d_out, int out_size, void* d_ws, size_t ws_size,
                              hipStream_t stream) {
    (void)in_sizes; (void)n_in; (void)out_size; (void)ws_size;
    const int Bz = 16, C = 512, D = 256, Nn = 128, T = 1024;
    const float* in[52];
    for (int i = 0; i < 52; ++i) in[i] = (const float*)d_in[i];

    // jax pytree (sorted dict keys) leaf order
    const float* F[2]       = { in[0],  in[1]  };
    const float* A_bar      = in[2];
    const float* B_bar      = in[3];
    const float* Cb[2]      = { in[4],  in[6]  };   // C_rgb.b, C_x.b
    const float* Cw[2]      = { in[5],  in[7]  };   // C_rgb.w, C_x.w
    const float* Db_[2]     = { in[8],  in[10] };
    const float* Dw_[2]     = { in[9],  in[11] };
    const float* Winb[2]    = { in[12], in[14] };
    const float* Winw[2]    = { in[13], in[15] };
    const float* dA_b1[2]   = { in[16], in[20] };
    const float* dA_w1[2]   = { in[17], in[21] };
    const float* dA_b2[2]   = { in[18], in[22] };
    const float* dA_w2[2]   = { in[19], in[23] };
    const float* dB_b1[2]   = { in[24], in[28] };
    const float* dB_w1[2]   = { in[25], in[29] };
    const float* dB_b2[2]   = { in[26], in[30] };
    const float* dB_w2[2]   = { in[27], in[31] };
    const float* lnob[2]    = { in[32], in[34] };
    const float* lnog[2]    = { in[33], in[35] };
    const float* outb[2]    = { in[36], in[38] };
    const float* outw[2]    = { in[37], in[39] };
    const float* pre_b1[2]  = { in[40], in[46] };
    const float* pre_b3[2]  = { in[41], in[47] };
    const float* pre_lnb[2] = { in[42], in[48] };
    const float* pre_lng[2] = { in[43], in[49] };
    const float* pre_w1[2]  = { in[44], in[50] };
    const float* pre_w3[2]  = { in[45], in[51] };

    // workspace layout (floats)
    float* ws = (float*)d_ws;
    const long long ZLNP = (long long)Bz * 1156 * D;     // padded pre buffer (also reused as y)
    const long long Z3   = (long long)Bz * T * D;
    const long long U    = (long long)Bz * T * Nn;
    long long off = 0;
    float* zlnp[2] = { ws + off, ws + off + ZLNP }; off += 2 * ZLNP;
    float* z3[2]   = { ws + off, ws + off + Z3   }; off += 2 * Z3;
    float* u[2]    = { ws + off, ws + off + U    }; off += 2 * U;
    float* h[2]    = { ws + off, ws + off + U    }; off += 2 * U;
    float* pooled[2] = { ws + off, ws + off + Bz * D }; off += 2 * Bz * D;
    float* Aco[2]  = { ws + off, ws + off + Bz * Nn }; off += 2 * Bz * Nn;
    float* Bco[2]  = { ws + off, ws + off + Bz * Nn }; off += 2 * Bz * Nn;
    float* y[2]    = { zlnp[0], zlnp[1] };               // alias: padded buffer dead after conv3

    // 1) zero padded buffers (borders must be 0)
    zero_kernel<<<2048, 256, 0, stream>>>(zlnp[0], 2 * ZLNP);

    for (int s = 0; s < 2; ++s) {
        // 2) 1x1 conv C->D, store into padded interior (row remap)
        GemmP p1 = { F[s], (long long)C * T, T, /*a_mode=*/1, 0,
                     pre_w1[s], C, 1, pre_b1[s],
                     zlnp[s], (long long)1156 * D, D, /*out_mode=*/2,
                     C, 0, 1 };
        gemm_wmma_bf16<<<dim3(T / BM, D / BN, Bz), 256, 0, stream>>>(p1);
        // 3) channel LN + SiLU in place
        ln_silu_kernel<<<Bz * T / 8, 256, 0, stream>>>(zlnp[s], pre_lng[s], pre_lnb[s]);
        // 4) 3x3 conv D->D as 9 WMMA-GEMM taps
        GemmP p3 = { zlnp[s], (long long)1156 * D, D, /*a_mode=*/2, 0,
                     pre_w3[s], D * 9, 9, pre_b3[s],
                     z3[s], (long long)T * D, D, 0,
                     D, 0, 9 };
        gemm_wmma_bf16<<<dim3(T / BM, D / BN, Bz), 256, 0, stream>>>(p3);
        // 5) pooled mean
        pool_kernel<<<Bz, 256, 0, stream>>>(z3[s], pooled[s]);
        // 6) delta heads
        head_kernel<<<Bz, 128, 0, stream>>>(pooled[s], dA_w1[s], dA_b1[s], dA_w2[s], dA_b2[s],
                                            A_bar, Aco[s], 1);
        head_kernel<<<Bz, 128, 0, stream>>>(pooled[s], dB_w1[s], dB_b1[s], dB_w2[s], dB_b2[s],
                                            B_bar, Bco[s], 0);
        // 7) u = z3 @ Win^T + b
        GemmP pw = { z3[s], (long long)T * D, D, 0, 0,
                     Winw[s], D, 1, Winb[s],
                     u[s], (long long)T * Nn, Nn, 0,
                     D, 0, 1 };
        gemm_wmma_bf16<<<dim3(T / BM, Nn / BN, Bz), 256, 0, stream>>>(pw);
        // 8) recurrence
        scan_kernel<<<Bz * Nn / 256, 256, 0, stream>>>(u[s], Aco[s], Bco[s], h[s]);
        // 9) y = h @ Ccross^T + b  (cross: stream 0 uses C_x, stream 1 uses C_rgb)
        GemmP pc = { h[s], (long long)T * Nn, Nn, 0, 0,
                     Cw[1 - s], Nn, 1, Cb[1 - s],
                     y[s], (long long)T * D, D, 0,
                     Nn, 0, 1 };
        gemm_wmma_bf16<<<dim3(T / BM, D / BN, Bz), 256, 0, stream>>>(pc);
        // 10) y += z3 @ Dmat^T + b
        GemmP pd = { z3[s], (long long)T * D, D, 0, 0,
                     Dw_[s], D, 1, Db_[s],
                     y[s], (long long)T * D, D, 0,
                     D, 1, 1 };
        gemm_wmma_bf16<<<dim3(T / BM, D / BN, Bz), 256, 0, stream>>>(pd);
        // 11) out conv 1x1 D->C, store transposed (NCHW) straight into d_out
        float* outp = (float*)d_out + (long long)s * Bz * C * T;
        GemmP po = { y[s], (long long)T * D, D, 0, 0,
                     outw[s], D, 1, outb[s],
                     outp, (long long)C * T, T, /*out_mode=*/1,
                     D, 0, 1 };
        gemm_wmma_bf16<<<dim3(T / BM, C / BN, Bz), 256, 0, stream>>>(po);
        // 12) final channel LN over C, in place on d_out
        ln_out_kernel<<<dim3(T / 256, Bz), 256, 0, stream>>>(outp, lnog[s], lnob[s]);
    }
}
